// MaskedTransformer_81801947120141
// MI455X (gfx1250) — matrix operand order repaired and checked by reading, not compiled, not run on hardware
//
#include <hip/hip_runtime.h>
#include <math.h>

#define B_   2
#define S_   1024
#define D_   1024
#define H_   16
#define DK_  64
#define DFF_ 4096

typedef __attribute__((ext_vector_type(16))) __bf16 v16bf;
typedef __attribute__((ext_vector_type(8)))  float  v8f;

union Frag {
  v16bf v;
  uint4 u[2];
};

__device__ __forceinline__ unsigned short f2bf(float f) {
  union { float f; unsigned int u; } c;
  c.f = f;
  unsigned int r = c.u + 0x7fffu + ((c.u >> 16) & 1u);  // round-to-nearest-even
  return (unsigned short)(r >> 16);
}

__device__ __forceinline__ float gelu_f(float x) {
  const float k = 0.7978845608028654f;  // sqrt(2/pi)
  return 0.5f * x * (1.0f + tanhf(k * (x + 0.044715f * x * x * x)));
}

// CDNA5 async global->LDS copy (ASYNCcnt-tracked), 16B per lane.
// VFLAT operand order: VDST (per-lane LDS byte address), VADDR (64-bit), SADDR=off.
__device__ __forceinline__ void async_load_b128(const void* gaddr, unsigned lds_off) {
  asm volatile("global_load_async_to_lds_b128 %0, %1, off"
               :: "v"(lds_off), "v"((unsigned long long)gaddr)
               : "memory");
}

// ---------------------------------------------------------------------------
// Batched bf16 WMMA GEMM:  C[z] = act( alpha * A[z] @ B[z]^T + bias )
//   A : [M][K] bf16 row-major, lda; per-z offset zb*aOffB + zh*aOffH
//   Bt: [N][K] bf16 row-major (B transposed), ldb
//   C : fp32 (Cf) and/or bf16 (Cb), ldc
// Block tile 128 x (WN*32) x 32, 128 threads (4 waves), wave tile 64 x (WN*16),
// 4*WN WMMA per K-step. LDS double-buffered, fed by global_load_async_to_lds.
// Requires M%128==0, N%(WN*32)==0, K%32==0 (true for all calls here).
// ---------------------------------------------------------------------------
template <int WN>  // n-frags per wave: 4 (BN=128) or 2 (BN=64)
__global__ __launch_bounds__(128)
void gemm_bf16_wmma(const unsigned short* __restrict__ A, int lda,
                    size_t aOffB, size_t aOffH,
                    const unsigned short* __restrict__ Bt, int ldb,
                    size_t bOffB, size_t bOffH,
                    float* __restrict__ Cf, unsigned short* __restrict__ Cb,
                    int ldc, size_t cOffB, size_t cOffH,
                    const float* __restrict__ bias,
                    int K, float alpha, int act, int zH) {
  constexpr int BN  = WN * 32;   // block N tile
  constexpr int CPT = BN / 32;   // B-tile b128 chunks per thread (2 or 4)
  __shared__ unsigned short As[2][128][40];  // +8 pad: 80B pitch, conflict-free b128
  __shared__ unsigned short Bs[2][BN][40];

  const int tid  = threadIdx.x;
  const int lane = tid & 31;
  const int wave = tid >> 5;
  const int wm = wave >> 1;   // 0..1 -> 64-row half
  const int wn = wave & 1;    // 0..1 -> (WN*16)-col half
  const int hi = lane >> 4;   // lane half
  const int lm = lane & 15;

  const int z  = blockIdx.z;
  const int zb = z / zH, zh = z % zH;
  A  += (size_t)zb * aOffB + (size_t)zh * aOffH;
  Bt += (size_t)zb * bOffB + (size_t)zh * bOffH;
  const size_t coff = (size_t)zb * cOffB + (size_t)zh * cOffH;

  const int m0 = blockIdx.y * 128;
  const int n0 = blockIdx.x * BN;

  v8f acc[4][WN];
  for (int i = 0; i < 4; ++i)
    for (int j = 0; j < WN; ++j)
      for (int e = 0; e < 8; ++e) acc[i][j][e] = 0.0f;

  const unsigned short* arow = A + (size_t)(m0 + tid) * lda;  // 32 bf16 / thread

  // Issue one K-stage of async global->LDS copies (4 + CPT instructions / lane).
  auto issue = [&](int kb, int buf) {
    const unsigned short* ap = arow + kb;
    const unsigned aoff = (unsigned)(size_t)&As[buf][tid][0];
    async_load_b128(ap,      aoff);
    async_load_b128(ap + 8,  aoff + 16);
    async_load_b128(ap + 16, aoff + 32);
    async_load_b128(ap + 24, aoff + 48);
    for (int i = 0; i < CPT; ++i) {
      const int c = tid * CPT + i;            // b128 chunk id over the B tile
      const int row = c >> 2, col = (c & 3) * 8;
      async_load_b128(Bt + (size_t)(n0 + row) * ldb + kb + col,
                      (unsigned)(size_t)&Bs[buf][row][col]);
    }
  };

  issue(0, 0);
  const int nk = K >> 5;
  for (int t = 0; t < nk; ++t) {
    const int buf = t & 1;
    if (t + 1 < nk) {
      issue((t + 1) << 5, buf ^ 1);
      // in-order completion: <= (4+CPT) outstanding  =>  stage t has landed
      asm volatile("s_wait_asynccnt %0" :: "n"(4 + CPT) : "memory");
    } else {
      asm volatile("s_wait_asynccnt 0" ::: "memory");
    }
    __syncthreads();  // whole tile visible to all waves

    // LDS -> fragments per ISA VGPR layouts
    Frag af[4], bf[WN];
    for (int mt = 0; mt < 4; ++mt) {
      const unsigned short* p = &As[buf][wm * 64 + mt * 16 + lm][hi * 8];
      af[mt].u[0] = *(const uint4*)(p);        // K {0..7}  (+8 for hi lanes)
      af[mt].u[1] = *(const uint4*)(p + 16);   // K {16..23}(+8 for hi lanes)
    }
    for (int nt = 0; nt < WN; ++nt) {
      const unsigned short* p = &Bs[buf][wn * (WN * 16) + nt * 16 + lm][hi * 8];
      bf[nt].u[0] = *(const uint4*)(p);        // K {0..7}  (+8 for hi lanes)
      bf[nt].u[1] = *(const uint4*)(p + 16);   // K {16..23}(+8 for hi lanes)
    }
    for (int mt = 0; mt < 4; ++mt)
      for (int nt = 0; nt < WN; ++nt)
        acc[mt][nt] = __builtin_amdgcn_wmma_f32_16x16x32_bf16(
            false, af[mt].v, false, bf[nt].v, (short)0, acc[mt][nt], false, false);

    __syncthreads();  // all waves done reading buf before it is refilled
  }

  // epilogue: scale, bias, activation, fp32/bf16 stores
  for (int mt = 0; mt < 4; ++mt) {
    for (int nt = 0; nt < WN; ++nt) {
      const int col = n0 + wn * (WN * 16) + nt * 16 + lm;
      const float bval = bias ? bias[col] : 0.0f;
      for (int e = 0; e < 8; ++e) {
        const int row = m0 + wm * 64 + mt * 16 + hi * 8 + e;
        float c = acc[mt][nt][e] * alpha + bval;
        if (act == 1) c = gelu_f(c);
        const size_t idx = coff + (size_t)row * ldc + col;
        if (Cf) Cf[idx] = c;
        if (Cb) Cb[idx] = f2bf(c);
      }
    }
  }
}

// ---------------------------------------------------------------------------
__global__ void cvt_f32_to_bf16(const float* __restrict__ src,
                                unsigned short* __restrict__ dst, size_t n) {
  size_t i = (size_t)blockIdx.x * blockDim.x + threadIdx.x;
  const size_t stride = (size_t)gridDim.x * blockDim.x;
  for (; i < n; i += stride) dst[i] = f2bf(src[i]);
}

// src [Kd][Nd] fp32  ->  dst [Nd][Kd] bf16 (B^T for the GEMM)
__global__ __launch_bounds__(256)
void transpose_cvt(const float* __restrict__ src, unsigned short* __restrict__ dst,
                   int Kd, int Nd) {
  __shared__ float tile[32][33];
  const int n0 = blockIdx.x * 32, k0 = blockIdx.y * 32;
  const int tx = threadIdx.x, ty = threadIdx.y;  // (32, 8)
  for (int r = 0; r < 4; ++r)
    tile[ty + r * 8][tx] = src[(size_t)(k0 + ty + r * 8) * Nd + (n0 + tx)];
  __syncthreads();
  for (int r = 0; r < 4; ++r)
    dst[(size_t)(n0 + ty + r * 8) * Kd + (k0 + tx)] = f2bf(tile[tx][ty + r * 8]);
}

// VT[b][h][d][j] = V[b*S+j][h*dk+d]   (bf16 -> bf16 per-head transpose)
__global__ void build_vt(const unsigned short* __restrict__ Vb,
                         unsigned short* __restrict__ VT) {
  const size_t idx = (size_t)blockIdx.x * blockDim.x + threadIdx.x;
  const size_t total = (size_t)B_ * H_ * DK_ * S_;
  if (idx >= total) return;
  const int j = (int)(idx % S_);
  const int d = (int)((idx / S_) % DK_);
  const int h = (int)((idx / ((size_t)S_ * DK_)) % H_);
  const int b = (int)(idx / ((size_t)S_ * DK_ * H_));
  VT[idx] = Vb[(size_t)(b * S_ + j) * D_ + h * DK_ + d];
}

// in-place masked softmax over last dim of P [B,H,S,S]; also emits bf16 copy
__global__ __launch_bounds__(256)
void softmax_mask(float* __restrict__ P, const float* __restrict__ mask,
                  unsigned short* __restrict__ Pb) {
  const int q = blockIdx.x, h = blockIdx.y, b = blockIdx.z;
  const int tid = threadIdx.x;
  float* row = P + (((size_t)(b * H_ + h)) * S_ + q) * S_;
  unsigned short* rowb = Pb + (((size_t)(b * H_ + h)) * S_ + q) * S_;
  const float* mrow = mask + ((size_t)b * S_ + q) * S_;

  float v[4];
  for (int t = 0; t < 4; ++t) v[t] = row[tid + t * 256] + mrow[tid + t * 256];

  __shared__ float redm[8], reds[8];
  float m = fmaxf(fmaxf(v[0], v[1]), fmaxf(v[2], v[3]));
  for (int o = 16; o > 0; o >>= 1) m = fmaxf(m, __shfl_xor(m, o, 32));
  if ((tid & 31) == 0) redm[tid >> 5] = m;
  __syncthreads();
  m = redm[0];
  for (int i = 1; i < 8; ++i) m = fmaxf(m, redm[i]);

  float s = 0.0f;
  for (int t = 0; t < 4; ++t) { v[t] = __expf(v[t] - m); s += v[t]; }
  for (int o = 16; o > 0; o >>= 1) s += __shfl_xor(s, o, 32);
  if ((tid & 31) == 0) reds[tid >> 5] = s;
  __syncthreads();
  s = 0.0f;
  for (int i = 0; i < 8; ++i) s += reds[i];
  const float inv = 1.0f / s;

  for (int t = 0; t < 4; ++t) {
    const int j = tid + t * 256;
    const float p = v[t] * inv;
    row[j] = p;
    rowb[j] = f2bf(p);
  }
}

// y = x + ( g*(x-mean)/(std+eps) + b ),  torch-style unbiased std (ddof=1)
__global__ __launch_bounds__(256)
void ln_residual(const float* __restrict__ X, const float* __restrict__ g,
                 const float* __restrict__ bb, float* __restrict__ outF,
                 unsigned short* __restrict__ outB) {
  const int r = blockIdx.x;
  const int tid = threadIdx.x;
  const float* x = X + (size_t)r * D_;

  float v[4], s = 0.0f, ss = 0.0f;
  for (int t = 0; t < 4; ++t) {
    v[t] = x[tid + t * 256];
    s += v[t];
    ss += v[t] * v[t];
  }
  for (int o = 16; o > 0; o >>= 1) {
    s += __shfl_xor(s, o, 32);
    ss += __shfl_xor(ss, o, 32);
  }
  __shared__ float rs[8], rss[8];
  if ((tid & 31) == 0) { rs[tid >> 5] = s; rss[tid >> 5] = ss; }
  __syncthreads();
  s = 0.0f; ss = 0.0f;
  for (int i = 0; i < 8; ++i) { s += rs[i]; ss += rss[i]; }

  const float mean = s / (float)D_;
  const float var  = (ss - (float)D_ * mean * mean) / (float)(D_ - 1);
  const float inv  = 1.0f / (sqrtf(fmaxf(var, 0.0f)) + 1e-6f);

  for (int t = 0; t < 4; ++t) {
    const int j = tid + t * 256;
    const float y = v[t] + (g[j] * (v[t] - mean) * inv + bb[j]);
    if (outF) outF[(size_t)r * D_ + j] = y;
    if (outB) outB[(size_t)r * D_ + j] = f2bf(y);
  }
}

// ---------------------------------------------------------------------------
extern "C" void kernel_launch(void* const* d_in, const int* in_sizes, int n_in,
                              void* d_out, int out_size, void* d_ws, size_t ws_size,
                              hipStream_t stream) {
  (void)in_sizes; (void)n_in; (void)out_size; (void)ws_size;

  const float* x    = (const float*)d_in[0];
  const float* mask = (const float*)d_in[1];
  const float* Wq   = (const float*)d_in[2];
  const float* bq   = (const float*)d_in[3];
  const float* Wk   = (const float*)d_in[4];
  const float* bk   = (const float*)d_in[5];
  const float* Wv   = (const float*)d_in[6];
  const float* bv   = (const float*)d_in[7];
  const float* Wo   = (const float*)d_in[8];
  const float* bo   = (const float*)d_in[9];
  const float* W1   = (const float*)d_in[10];
  const float* b1   = (const float*)d_in[11];
  const float* W2   = (const float*)d_in[12];
  const float* b2   = (const float*)d_in[13];
  const float* g1   = (const float*)d_in[14];
  const float* be1  = (const float*)d_in[15];
  const float* g2   = (const float*)d_in[16];
  const float* be2  = (const float*)d_in[17];

  float* outx  = (float*)d_out;                  // [B,S,D]
  float* pattn = outx + (size_t)B_ * S_ * D_;    // [B,H,S,S]

  char* ws = (char*)d_ws;
  size_t off = 0;
  auto alloc = [&](size_t bytes) -> char* {
    char* p = ws + off;
    off = (off + bytes + 255) & ~(size_t)255;
    return p;
  };

  const size_t NT = (size_t)B_ * S_ * D_;  // 2M tokens*dim
  unsigned short* xb   = (unsigned short*)alloc(NT * 2);
  unsigned short* WqT  = (unsigned short*)alloc((size_t)D_ * D_ * 2);
  unsigned short* WkT  = (unsigned short*)alloc((size_t)D_ * D_ * 2);
  unsigned short* WvT  = (unsigned short*)alloc((size_t)D_ * D_ * 2);
  unsigned short* WoT  = (unsigned short*)alloc((size_t)D_ * D_ * 2);
  unsigned short* W1T  = (unsigned short*)alloc((size_t)DFF_ * D_ * 2);  // [DFF][D]
  unsigned short* W2T  = (unsigned short*)alloc((size_t)D_ * DFF_ * 2);  // [D][DFF]
  unsigned short* Qb   = (unsigned short*)alloc(NT * 2);
  unsigned short* Kb   = (unsigned short*)alloc(NT * 2);
  unsigned short* Vb   = (unsigned short*)alloc(NT * 2);
  unsigned short* VTb  = (unsigned short*)alloc(NT * 2);                  // [B,H,dk,S]
  unsigned short* Pbuf = (unsigned short*)alloc((size_t)B_ * H_ * S_ * S_ * 2);
  unsigned short* Ob   = (unsigned short*)alloc(NT * 2);
  float*          a32  = (float*)alloc(NT * 4);
  unsigned short* x1b  = (unsigned short*)alloc(NT * 2);
  unsigned short* h1b  = (unsigned short*)alloc((size_t)B_ * S_ * DFF_ * 2);
  float*          fbuf = (float*)alloc(NT * 4);

  const size_t SD = (size_t)S_ * D_, SS = (size_t)S_ * S_;

  // 1) precision conversions
  cvt_f32_to_bf16<<<2048, 256, 0, stream>>>(x, xb, NT);
  transpose_cvt<<<dim3(D_ / 32, D_ / 32), dim3(32, 8), 0, stream>>>(Wq, WqT, D_, D_);
  transpose_cvt<<<dim3(D_ / 32, D_ / 32), dim3(32, 8), 0, stream>>>(Wk, WkT, D_, D_);
  transpose_cvt<<<dim3(D_ / 32, D_ / 32), dim3(32, 8), 0, stream>>>(Wv, WvT, D_, D_);
  transpose_cvt<<<dim3(D_ / 32, D_ / 32), dim3(32, 8), 0, stream>>>(Wo, WoT, D_, D_);
  transpose_cvt<<<dim3(DFF_ / 32, D_ / 32), dim3(32, 8), 0, stream>>>(W1, W1T, D_, DFF_);
  transpose_cvt<<<dim3(D_ / 32, DFF_ / 32), dim3(32, 8), 0, stream>>>(W2, W2T, DFF_, D_);

  // 2) QKV projections (bf16 out)
  gemm_bf16_wmma<4><<<dim3(D_ / 128, (B_ * S_) / 128, 1), 128, 0, stream>>>(
      xb, D_, 0, 0, WqT, D_, 0, 0, nullptr, Qb, D_, 0, 0, bq, D_, 1.0f, 0, 1);
  gemm_bf16_wmma<4><<<dim3(D_ / 128, (B_ * S_) / 128, 1), 128, 0, stream>>>(
      xb, D_, 0, 0, WkT, D_, 0, 0, nullptr, Kb, D_, 0, 0, bk, D_, 1.0f, 0, 1);
  gemm_bf16_wmma<4><<<dim3(D_ / 128, (B_ * S_) / 128, 1), 128, 0, stream>>>(
      xb, D_, 0, 0, WvT, D_, 0, 0, nullptr, Vb, D_, 0, 0, bv, D_, 1.0f, 0, 1);
  build_vt<<<(unsigned)((NT + 255) / 256), 256, 0, stream>>>(Vb, VTb);

  // 3) scores = (Q @ K^T) / sqrt(dk)  -> fp32 p_attn region of d_out
  gemm_bf16_wmma<4><<<dim3(S_ / 128, S_ / 128, B_ * H_), 128, 0, stream>>>(
      Qb, D_, SD, DK_, Kb, D_, SD, DK_,
      pattn, nullptr, S_, (size_t)H_ * SS, SS, nullptr,
      DK_, 0.125f, 0, H_);

  // 4) masked softmax (fp32 in place + bf16 copy for PV)
  softmax_mask<<<dim3(S_, H_, B_), 256, 0, stream>>>(pattn, mask, Pbuf);

  // 5) o = P @ V  (N=64 -> narrow-tile variant; bf16 out into head columns)
  gemm_bf16_wmma<2><<<dim3(DK_ / 64, S_ / 128, B_ * H_), 128, 0, stream>>>(
      Pbuf, S_, (size_t)H_ * SS, SS, VTb, S_, (size_t)H_ * DK_ * S_, (size_t)DK_ * S_,
      nullptr, Ob, D_, SD, DK_, nullptr, S_, 1.0f, 0, H_);

  // 6) a = o @ Wo + bo  (fp32 for layernorm)
  gemm_bf16_wmma<4><<<dim3(D_ / 128, (B_ * S_) / 128, 1), 128, 0, stream>>>(
      Ob, D_, 0, 0, WoT, D_, 0, 0, a32, nullptr, D_, 0, 0, bo, D_, 1.0f, 0, 1);

  // 7) x1 = a + LN(a)  (bf16 for FFN)
  ln_residual<<<B_ * S_, 256, 0, stream>>>(a32, g1, be1, nullptr, x1b);

  // 8) h1 = gelu(x1 @ W1 + b1)  (bf16)
  gemm_bf16_wmma<4><<<dim3(DFF_ / 128, (B_ * S_) / 128, 1), 128, 0, stream>>>(
      x1b, D_, 0, 0, W1T, D_, 0, 0, nullptr, h1b, DFF_, 0, 0, b1, D_, 1.0f, 1, 1);

  // 9) f = h1 @ W2 + b2  (fp32)
  gemm_bf16_wmma<4><<<dim3(D_ / 128, (B_ * S_) / 128, 1), 128, 0, stream>>>(
      h1b, DFF_, 0, 0, W2T, DFF_, 0, 0, fbuf, nullptr, D_, 0, 0, b2, DFF_, 1.0f, 0, 1);

  // 10) out = f + LN(f)  (fp32 -> d_out)
  ln_residual<<<B_ * S_, 256, 0, stream>>>(fbuf, g2, be2, outx, nullptr);
}
